// MemoryBank_50319836840644
// MI455X (gfx1250) — matrix operand order repaired and checked by reading, not compiled, not run on hardware
//
#include <hip/hip_runtime.h>
#include <hip/hip_bf16.h>
#include <math.h>

typedef _Float16 half_t;
typedef __attribute__((ext_vector_type(16))) _Float16 v16h;
typedef __attribute__((ext_vector_type(8)))  _Float16 v8h;
typedef __attribute__((ext_vector_type(4)))  _Float16 v4h;
typedef __attribute__((ext_vector_type(8)))  float    v8f;
typedef __attribute__((ext_vector_type(4)))  int      v4i;

// CDNA5 async global->LDS copy path (ASYNCcnt-tracked, CDNA5 ISA).
// Builtin signature (from hipcc diagnostic): (int4 AS1* src, int4 AS3* dst, imm, imm)
#if defined(__AMDGCN__) && \
    __has_builtin(__builtin_amdgcn_global_load_async_to_lds_b128) && \
    __has_builtin(__builtin_amdgcn_s_wait_asynccnt)
#define USE_ASYNC_LDS 1
#else
#define USE_ASYNC_LDS 0
#endif

#if USE_ASYNC_LDS
typedef v4i __attribute__((address_space(1)))* gas_v4i_t;
typedef v4i __attribute__((address_space(3)))* las_v4i_t;
// AS(1) pointers are 64-bit: reinterpret via integer.
__device__ inline gas_v4i_t as_global_v4i(const void* p) {
    return (gas_v4i_t)(size_t)p;
}
// AS(3) pointers are 32-bit; generic LDS pointer's low 32 bits are the LDS
// byte offset.
__device__ inline las_v4i_t as_lds_v4i(void* p) {
    return (las_v4i_t)(unsigned int)(size_t)p;
}
#endif

// ---------------------------------------------------------------------------
// WMMA fragment helpers (wave32; A/B in contiguous k order 16*(lane/16)+s, C/D per ISA)
// ---------------------------------------------------------------------------

// A-matrix 16x32 f16, source row-major [16 rows x ld], tile base pre-offset.
// lane<16 : row=lane, K 0..15; lane>=16 : row=lane-16, K 16..31
__device__ inline v16h load_frag_a_f16(const half_t* __restrict__ src, int ld, int lane) {
    const int r  = lane & 15;
    const int hi = lane >> 4;
    const half_t* p = src + (size_t)r * ld + hi * 16;
    union { v16h v; v8h h[2]; } u;
    u.h[0] = *(const v8h*)(p);
    u.h[1] = *(const v8h*)(p + 8);
    return u.v;
}

// Same fragment but from an f32 source (convert on the fly).
__device__ inline v16h load_frag_a_f32(const float* __restrict__ src, int ld, int lane) {
    const int r  = lane & 15;
    const int hi = lane >> 4;
    const float* p = src + (size_t)r * ld + hi * 16;
    union { v16h v; v8h h[2]; } u;
#pragma unroll
    for (int j = 0; j < 2; ++j) {
        const float4 f0 = *(const float4*)(p + j * 8);
        const float4 f1 = *(const float4*)(p + j * 8 + 4);
        v8h t;
        t[0] = (_Float16)f0.x; t[1] = (_Float16)f0.y;
        t[2] = (_Float16)f0.z; t[3] = (_Float16)f0.w;
        t[4] = (_Float16)f1.x; t[5] = (_Float16)f1.y;
        t[6] = (_Float16)f1.z; t[7] = (_Float16)f1.w;
        u.h[j] = t;
    }
    return u.v;
}

// B-matrix 32x16 f16 (K x N) where B = Y^T and Y is row-major [16 rows x ld]:
// lane<16 : col=lane, K=0..15 ; lane>=16 : col=lane-16, K=16..31  (32B contig/lane)
__device__ inline v16h load_frag_b_global(const half_t* __restrict__ srcY, int ld, int lane) {
    const int n  = lane & 15;
    const int hi = lane >> 4;
    return *(const v16h*)(srcY + (size_t)n * ld + hi * 16);
}

#if USE_ASYNC_LDS
// LDS-staged B tile: 128 rows x 32 halves, row stride 40 halves (80 B).
// lane n reads bytes (col+n)*80 + hi*32: n*20 mod 64 banks all distinct -> conflict-free.
#define BS_STRIDE 40
__device__ inline v16h load_frag_b_lds(const half_t* base, int localcol, int lane) {
    const int n  = lane & 15;
    const int hi = lane >> 4;
    return *(const v16h*)(base + (localcol + n) * BS_STRIDE + hi * 16);
}
#endif

// C/D 16x16 f32: VGPR v <-> row v + 8*(lane>=16), col = lane&15
__device__ inline void store_frag_c(float* __restrict__ dst, int ld, int lane, v8f acc) {
    const int n  = lane & 15;
    const int hi = lane >> 4;
    float* p = dst + (size_t)(hi * 8) * ld + n;
#pragma unroll
    for (int v = 0; v < 8; ++v) p[(size_t)v * ld] = acc[v];
}

// ---------------------------------------------------------------------------
// Kernel 1: row L2-normalize f32 -> f16, optional transposed copy
// ---------------------------------------------------------------------------
template <bool WRITE_T>
__global__ __launch_bounds__(256) void l2norm_rows(const float* __restrict__ src,
                                                   half_t* __restrict__ dst,
                                                   half_t* __restrict__ dstT,
                                                   int D, int nrows) {
    const int row = blockIdx.x;
    const int t   = threadIdx.x;
    __shared__ float red[256];

    const float4 x = ((const float4*)(src + (size_t)row * D))[t];
    red[t] = x.x * x.x + x.y * x.y + x.z * x.z + x.w * x.w;
    __syncthreads();
#pragma unroll
    for (int s = 128; s > 0; s >>= 1) {
        if (t < s) red[t] += red[t + s];
        __syncthreads();
    }
    const float scale = 1.0f / fmaxf(sqrtf(red[0]), 1e-12f);

    v4h h;
    h[0] = (_Float16)(x.x * scale);
    h[1] = (_Float16)(x.y * scale);
    h[2] = (_Float16)(x.z * scale);
    h[3] = (_Float16)(x.w * scale);
    *(v4h*)(dst + (size_t)row * D + t * 4) = h;

    if constexpr (WRITE_T) {
        const int c = t * 4;
#pragma unroll
        for (int e = 0; e < 4; ++e)
            dstT[(size_t)(c + e) * nrows + row] = h[e];
    }
}

// ---------------------------------------------------------------------------
// Kernel 2: NT GEMM with WMMA:  C[Mr,Nc] = A[Mr,K] * Y[Nc,K]^T   (f32 accum)
// Block = 256 threads (8 waves), block tile 128x128, wave tile 32x64.
// B tile staged to LDS via async global->LDS copies, double-buffered.
// ---------------------------------------------------------------------------
template <bool A_F32>
__global__ __launch_bounds__(256) void gemm_nt_wmma(const void* __restrict__ Aptr,
                                                    const half_t* __restrict__ Y,
                                                    float* __restrict__ C,
                                                    int Mr, int Nc, int K) {
    const int lane = threadIdx.x & 31;
    const int wave = threadIdx.x >> 5;
    const int wm   = wave & 3;   // 4 waves along M
    const int wn   = wave >> 2;  // 2 waves along N
    const int row0 = blockIdx.y * 128 + wm * 32;
    const int col0 = blockIdx.x * 128 + wn * 64;

    const half_t* Ah = (const half_t*)Aptr;
    const float*  Af = (const float*)Aptr;

    v8f acc[2][4];
#pragma unroll
    for (int i = 0; i < 2; ++i)
#pragma unroll
        for (int j = 0; j < 4; ++j) acc[i][j] = {};

#if USE_ASYNC_LDS
    __shared__ half_t Bs[2][128 * BS_STRIDE];  // 20 KB (<< 320 KB/WGP)
    const int t = threadIdx.x;
    // Stage 128x32-half B tile: 512 chunks of 16 B; 2 chunks per thread.
    auto stage = [&](int k0, int buf) {
#pragma unroll
        for (int i = 0; i < 2; ++i) {
            const int chunk = t + i * 256;
            const int rrow  = chunk >> 2;   // 0..127 tile row
            const int cc    = chunk & 3;    // 16B chunk within the 64B row
            const half_t* g = Y + (size_t)(blockIdx.x * 128 + rrow) * K + k0 + cc * 8;
            half_t* l = &Bs[buf][rrow * BS_STRIDE + cc * 8];
            __builtin_amdgcn_global_load_async_to_lds_b128(as_global_v4i(g),
                                                           as_lds_v4i(l), 0, 0);
        }
    };
    stage(0, 0);
    int buf = 0;
#endif

    for (int k0 = 0; k0 < K; k0 += 32) {
#if USE_ASYNC_LDS
        if (k0 + 32 < K) {
            stage(k0 + 32, buf ^ 1);
            __builtin_amdgcn_s_wait_asynccnt(2);  // current tile landed (in-order)
        } else {
            __builtin_amdgcn_s_wait_asynccnt(0);
        }
        __syncthreads();
#endif
        // Prefetch the A stream two K-tiles ahead (global_prefetch_b8).
        if (k0 + 64 < K) {
            if constexpr (A_F32)
                __builtin_prefetch(Af + (size_t)(row0 + (lane & 15)) * K + k0 + 64, 0, 0);
            else
                __builtin_prefetch(Ah + (size_t)(row0 + (lane & 15)) * K + k0 + 64, 0, 0);
        }

        v16h a[2], b[4];
#pragma unroll
        for (int i = 0; i < 2; ++i) {
            if constexpr (A_F32)
                a[i] = load_frag_a_f32(Af + (size_t)(row0 + i * 16) * K + k0, K, lane);
            else
                a[i] = load_frag_a_f16(Ah + (size_t)(row0 + i * 16) * K + k0, K, lane);
        }
#pragma unroll
        for (int j = 0; j < 4; ++j) {
#if USE_ASYNC_LDS
            b[j] = load_frag_b_lds(&Bs[buf][0], wn * 64 + j * 16, lane);
#else
            b[j] = load_frag_b_global(Y + (size_t)(col0 + j * 16) * K + k0, K, lane);
#endif
        }

#pragma unroll
        for (int i = 0; i < 2; ++i)
#pragma unroll
            for (int j = 0; j < 4; ++j)
                acc[i][j] = __builtin_amdgcn_wmma_f32_16x16x32_f16(
                    false, a[i], false, b[j], (short)0, acc[i][j], false, false);

#if USE_ASYNC_LDS
        buf ^= 1;
        __syncthreads();  // all waves done reading before this buffer is re-staged
#endif
    }

#pragma unroll
    for (int i = 0; i < 2; ++i)
#pragma unroll
        for (int j = 0; j < 4; ++j)
            store_frag_c(C + (size_t)(row0 + i * 16) * Nc + (col0 + j * 16), Nc, lane,
                         acc[i][j]);
}

// ---------------------------------------------------------------------------
// Kernel 3: in-place row softmax, row length 8192, block 256, 32 elems/thread
// ---------------------------------------------------------------------------
__global__ __launch_bounds__(256) void softmax_rows(float* __restrict__ attn, int ncols) {
    const int row = blockIdx.x;
    const int t   = threadIdx.x;
    float* p = attn + (size_t)row * ncols;
    __shared__ float red[256];

    float vals[32];
    float m = -INFINITY;
#pragma unroll
    for (int i = 0; i < 8; ++i) {
        const float4 v = ((const float4*)p)[t + i * 256];
        vals[i * 4 + 0] = v.x; vals[i * 4 + 1] = v.y;
        vals[i * 4 + 2] = v.z; vals[i * 4 + 3] = v.w;
        m = fmaxf(m, fmaxf(fmaxf(v.x, v.y), fmaxf(v.z, v.w)));
    }
    red[t] = m;
    __syncthreads();
#pragma unroll
    for (int s = 128; s > 0; s >>= 1) {
        if (t < s) red[t] = fmaxf(red[t], red[t + s]);
        __syncthreads();
    }
    const float rowmax = red[0];
    __syncthreads();

    float sum = 0.0f;
#pragma unroll
    for (int i = 0; i < 32; ++i) {
        vals[i] = __expf(vals[i] - rowmax);
        sum += vals[i];
    }
    red[t] = sum;
    __syncthreads();
#pragma unroll
    for (int s = 128; s > 0; s >>= 1) {
        if (t < s) red[t] += red[t + s];
        __syncthreads();
    }
    const float inv = 1.0f / red[0];

#pragma unroll
    for (int i = 0; i < 8; ++i) {
        float4 w;
        w.x = vals[i * 4 + 0] * inv; w.y = vals[i * 4 + 1] * inv;
        w.z = vals[i * 4 + 2] * inv; w.w = vals[i * 4 + 3] * inv;
        ((float4*)p)[t + i * 256] = w;
    }
}

// ---------------------------------------------------------------------------
// Launch
// ---------------------------------------------------------------------------
extern "C" void kernel_launch(void* const* d_in, const int* in_sizes, int n_in,
                              void* d_out, int out_size, void* d_ws, size_t ws_size,
                              hipStream_t stream) {
    const int D = 1024;
    const int B = in_sizes[0] / D;  // 4096 query rows
    const int M = in_sizes[1] / D;  // 8192 memory rows

    const float* query  = (const float*)d_in[0];  // [B, D]
    const float* memory = (const float*)d_in[1];  // [M, D]

    float* out_feat = (float*)d_out;                  // [B, D]
    float* out_attn = (float*)d_out + (size_t)B * D;  // [B, M]

    // workspace: q_hat [B*D] f16, m_hat [M*D] f16, m_hat^T [D*M] f16  (~40 MB)
    half_t* qn    = (half_t*)d_ws;
    half_t* memn  = qn + (size_t)B * D;
    half_t* memnT = memn + (size_t)M * D;

    l2norm_rows<true><<<M, 256, 0, stream>>>(memory, memn, memnT, D, M);
    l2norm_rows<false><<<B, 256, 0, stream>>>(query, qn, nullptr, D, B);

    // sim = q_hat @ m_hat^T -> attn region (f32), K = D
    dim3 g1(M / 128, B / 128);
    gemm_nt_wmma<false><<<g1, 256, 0, stream>>>(qn, memn, out_attn, B, M, D);

    softmax_rows<<<B, 256, 0, stream>>>(out_attn, M);

    // feat = attn @ m_hat = attn @ (m_hat^T)^T, A = f32 on the fly, K = M
    dim3 g2(D / 128, B / 128);
    gemm_nt_wmma<true><<<g2, 256, 0, stream>>>(out_attn, memnT, out_feat, B, D, M);
}
